// BatGraphAttentionLayer_80556406604049
// MI455X (gfx1250) — compile-verified, matrix operand order checked by reading, not run on hardware
//
#include <hip/hip_runtime.h>
#include <hip/hip_bf16.h>
#include <cstdint>

typedef __attribute__((ext_vector_type(2)))  float  v2f;
typedef __attribute__((ext_vector_type(8)))  float  v8f;
typedef __attribute__((ext_vector_type(16))) __bf16 v16bf;

#define GAT_ALPHA 0.4f
#define GAT_NEGINF -9.0e15f
#define BSZ  8
#define NN   2048
#define FIN  256
#define FOUT 128
#define PSTRIDE (NN + 8)   // padded bf16 row stride: 4112B = 257*16B -> conflict-free b128 LDS reads

// LDS carve offsets for kernel 2 (bytes)
#define OFF_P    0
#define OFF_S2   65792
#define OFF_RED  (OFF_S2 + 8192)
#define OFF_S1   (OFF_RED + 1024)
#define OFF_MX   (OFF_S1 + 64)
#define OFF_RS   (OFF_MX + 64)
#define OFF_HBUF 75200          // 16B aligned
#define HROW     40             // staged h row stride in bf16 elems (80B: 20n mod 64 distinct -> no conflicts)
#define HBUFB    10240          // one buffer: 128 rows * 80B
#define SMEM_SZ  (OFF_HBUF + 2 * HBUFB)   // 95,680 B

__device__ __forceinline__ unsigned short f2bf(float f) {
  unsigned int u = __float_as_uint(f);
  u += 0x7FFFu + ((u >> 16) & 1u);   // round-to-nearest-even
  return (unsigned short)(u >> 16);
}

// CDNA5 async DMA: LDS[vdst] = MEM[vaddr], 16 bytes, tracked by ASYNCcnt.
__device__ __forceinline__ void async_load_b128(unsigned lds_off, const void* gptr) {
  asm volatile("global_load_async_to_lds_b128 %0, %1, off"
               :: "v"(lds_off), "v"((unsigned long long)(uintptr_t)gptr)
               : "memory");
}
__device__ __forceinline__ void wait_asynccnt0() {
  asm volatile("s_wait_asynccnt 0x0" ::: "memory");
}

// ---------------- Kernel 1: h = input @ W^T + b  (fp32 WMMA 16x16x4) ----------------
__global__ __launch_bounds__(256) void gat_gemm1(
    const float* __restrict__ inp, const float* __restrict__ Wm,
    const float* __restrict__ bias, const float* __restrict__ avec,
    unsigned short* __restrict__ hT,
    float* __restrict__ ps1, float* __restrict__ ps2)
{
  const int lane = threadIdx.x & 31;
  const int lo = lane & 15, hi = lane >> 4;
  const int wid   = (blockIdx.x * blockDim.x + threadIdx.x) >> 5;
  const int mtile = wid >> 3;            // 1024 tiles over B*NN rows
  const int nt    = wid & 7;             // 8 tiles over FOUT
  const int mbase = mtile << 4;
  const int nb    = nt << 4;
  const int mrow  = mbase + lo;          // A: lanes 0-15 and 16-31 both map M=0..15
  const int ncol  = nb + lo;             // B: lane -> N

  const float* arow = inp + (size_t)mrow * FIN;
  const float* wrow = Wm  + (size_t)ncol * FIN;

  v8f acc = {};
  #pragma unroll 8
  for (int kb = 0; kb < FIN; kb += 4) {
    const int k0 = kb + (hi << 1);       // 32-bit A layout: VGPR0=K0/K2, VGPR1=K1/K3
    v2f a  = *(const v2f*)(arow + k0);
    v2f bm = *(const v2f*)(wrow + k0);
    acc = __builtin_amdgcn_wmma_f32_16x16x4_f32(false, a, false, bm, (short)0, acc,
                                                false, false);
  }
  const float bv = bias[ncol];
  #pragma unroll
  for (int r = 0; r < 8; ++r) acc[r] += bv;

  // hT bf16 [B][FOUT][NN]; lane holds rows mbase+8*hi+r at column ncol -> one b128 store
  const int bb   = mbase >> 11;
  const int rowb = (mbase & (NN - 1)) + (hi << 3);
  uint4 pk;
  pk.x = (unsigned)f2bf(acc[0]) | ((unsigned)f2bf(acc[1]) << 16);
  pk.y = (unsigned)f2bf(acc[2]) | ((unsigned)f2bf(acc[3]) << 16);
  pk.z = (unsigned)f2bf(acc[4]) | ((unsigned)f2bf(acc[5]) << 16);
  pk.w = (unsigned)f2bf(acc[6]) | ((unsigned)f2bf(acc[7]) << 16);
  *(uint4*)(hT + ((size_t)(bb * FOUT + ncol)) * NN + rowb) = pk;

  // deterministic per-ntile partial scores (16-lane shuffle reduction)
  const float a1v = avec[ncol];
  const float a2v = avec[FOUT + ncol];
  #pragma unroll
  for (int r = 0; r < 8; ++r) {
    float t1 = acc[r] * a1v;
    float t2 = acc[r] * a2v;
    #pragma unroll
    for (int mm = 8; mm >= 1; mm >>= 1) {
      t1 += __shfl_xor(t1, mm, 32);
      t2 += __shfl_xor(t2, mm, 32);
    }
    if (lo == 0) {
      const int mg = mbase + (hi << 3) + r;
      ps1[nt * (BSZ * NN) + mg] = t1;
      ps2[nt * (BSZ * NN) + mg] = t2;
    }
  }
}

// ---------------- Kernel 1b: reduce partials -> s1, s2 ----------------
__global__ __launch_bounds__(256) void gat_scores(
    const float* __restrict__ ps1, const float* __restrict__ ps2,
    float* __restrict__ s1, float* __restrict__ s2)
{
  const int i = blockIdx.x * 256 + threadIdx.x;
  if (i >= BSZ * NN) return;
  float a = 0.f, b = 0.f;
  #pragma unroll
  for (int t = 0; t < 8; ++t) {
    a += ps1[t * (BSZ * NN) + i];
    b += ps2[t * (BSZ * NN) + i];
  }
  s1[i] = a; s2[i] = b;
}

// ------- Kernel 2: fused masked-softmax + attn@h (bf16 WMMA, async-LDS staging) + ELU -------
__global__ __launch_bounds__(256) void gat_attn(
    const int* __restrict__ adj, const unsigned short* __restrict__ hT,
    const float* __restrict__ s1g, const float* __restrict__ s2g,
    float* __restrict__ out)
{
  extern __shared__ char smem[];
  unsigned short* P  = (unsigned short*)(smem + OFF_P);
  float* s2s = (float*)(smem + OFF_S2);
  float* red = (float*)(smem + OFF_RED);
  float* s1s = (float*)(smem + OFF_S1);
  float* mx  = (float*)(smem + OFF_MX);
  float* rs  = (float*)(smem + OFF_RS);

  const int tid = threadIdx.x;
  const int bb  = blockIdx.x >> 7;           // 128 row-tiles per batch
  const int i0  = (blockIdx.x & 127) << 4;

  for (int j = tid; j < NN; j += 256) s2s[j] = s2g[bb * NN + j];
  if (tid < 16) s1s[tid] = s1g[bb * NN + i0 + tid];
  __syncthreads();

  const int row = tid >> 4;
  const int sub = tid & 15;
  const int4* adjrow4 = (const int4*)(adj + ((size_t)(bb * NN + i0 + row)) * NN);
  const float s1v = s1s[row];

  // Pass A: row max of masked leaky-relu logits (int4 adj / float4 s2)
  float lmax = GAT_NEGINF;
  for (int c = 0; c < 32; ++c) {
    const int idx = (c << 4) + sub;                    // int4 index; j = idx*4
    if (c < 24) __builtin_prefetch(adjrow4 + idx + 128, 0, 0);  // global_prefetch_b8
    const int4  av = adjrow4[idx];
    const float4 sv = *(const float4*)(s2s + (idx << 2));
    float e0 = s1v + sv.x; e0 = e0 > 0.f ? e0 : GAT_ALPHA * e0; e0 = av.x > 0 ? e0 : GAT_NEGINF;
    float e1 = s1v + sv.y; e1 = e1 > 0.f ? e1 : GAT_ALPHA * e1; e1 = av.y > 0 ? e1 : GAT_NEGINF;
    float e2 = s1v + sv.z; e2 = e2 > 0.f ? e2 : GAT_ALPHA * e2; e2 = av.z > 0 ? e2 : GAT_NEGINF;
    float e3 = s1v + sv.w; e3 = e3 > 0.f ? e3 : GAT_ALPHA * e3; e3 = av.w > 0 ? e3 : GAT_NEGINF;
    lmax = fmaxf(lmax, fmaxf(fmaxf(e0, e1), fmaxf(e2, e3)));
  }
  red[tid] = lmax;
  __syncthreads();
  if (tid < 16) {
    float m = GAT_NEGINF;
    #pragma unroll
    for (int k = 0; k < 16; ++k) m = fmaxf(m, red[(tid << 4) + k]);
    mx[tid] = m;
  }
  __syncthreads();

  // Pass B: unnormalized exp -> bf16 P tile in LDS (uint2 packed stores) + row sums
  const float mrow = mx[row];
  float lsum = 0.f;
  for (int c = 0; c < 32; ++c) {
    const int idx = (c << 4) + sub;
    const int4  av = adjrow4[idx];
    const float4 sv = *(const float4*)(s2s + (idx << 2));
    float e0 = s1v + sv.x; e0 = e0 > 0.f ? e0 : GAT_ALPHA * e0; e0 = av.x > 0 ? e0 : GAT_NEGINF;
    float e1 = s1v + sv.y; e1 = e1 > 0.f ? e1 : GAT_ALPHA * e1; e1 = av.y > 0 ? e1 : GAT_NEGINF;
    float e2 = s1v + sv.z; e2 = e2 > 0.f ? e2 : GAT_ALPHA * e2; e2 = av.z > 0 ? e2 : GAT_NEGINF;
    float e3 = s1v + sv.w; e3 = e3 > 0.f ? e3 : GAT_ALPHA * e3; e3 = av.w > 0 ? e3 : GAT_NEGINF;
    const float p0 = __expf(e0 - mrow), p1 = __expf(e1 - mrow);
    const float p2 = __expf(e2 - mrow), p3 = __expf(e3 - mrow);
    lsum += (p0 + p1) + (p2 + p3);
    uint2 pk;
    pk.x = (unsigned)f2bf(p0) | ((unsigned)f2bf(p1) << 16);
    pk.y = (unsigned)f2bf(p2) | ((unsigned)f2bf(p3) << 16);
    *(uint2*)(P + row * PSTRIDE + (idx << 2)) = pk;
  }
  red[tid] = lsum;
  __syncthreads();
  if (tid < 16) {
    float s = 0.f;
    #pragma unroll
    for (int k = 0; k < 16; ++k) s += red[(tid << 4) + k];
    rs[tid] = 1.0f / s;
  }
  __syncthreads();

  // Pass C: D(16x16 per wave) = P(16xNN) @ h(NNxFOUT), 64 K-steps of 32,
  // B-operand h-tile (128n x 32k bf16 = 8KB) double-buffered in LDS via async DMA.
  const int w = tid >> 5, lane = tid & 31;
  const int lo = lane & 15, hi = lane >> 4;
  const int nb = w << 4;
  const int ncol = nb + lo;
  const unsigned short* Pr = P + lo * PSTRIDE;

  // async staging: thread t moves 32B of row n = t>>1, k-half = t&1
  const int sn = tid >> 1, shalf = tid & 1;
  const unsigned short* gsrc0 = hT + ((size_t)(bb * FOUT + sn)) * NN + shalf * 16;
  const unsigned ldst0 = (unsigned)(uintptr_t)(smem + OFF_HBUF) + sn * (HROW * 2) + shalf * 32;

  // prologue: stage k-step 0 into buffer 0
  async_load_b128(ldst0, gsrc0);
  async_load_b128(ldst0 + 16, gsrc0 + 8);

  union ABu { uint4 q[2]; v16bf v; };
  v8f acc = {};
  for (int ks = 0; ks < NN / 32; ++ks) {
    const int kb  = ks << 5;
    const int cur = ks & 1;
    wait_asynccnt0();      // my chunks of buf[cur] landed
    __syncthreads();       // everyone's landed; prior reads of buf[cur^1] fully consumed

    if (ks + 1 < NN / 32) {
      const unsigned short* gs = gsrc0 + (kb + 32);
      const unsigned ld = ldst0 + (cur ^ 1) * HBUFB;
      async_load_b128(ld, gs);
      async_load_b128(ld + 16, gs + 8);
    }

    const unsigned short* Hr = (const unsigned short*)(smem + OFF_HBUF + cur * HBUFB)
                               + ncol * HROW;
    // 16-bit operand layout: lanes hi=0 hold K {0..7,16..23}, hi=1 hold K {8..15,24..31}
    ABu a, b;
    a.q[0] = *(const uint4*)(Pr + kb + (hi << 3));
    a.q[1] = *(const uint4*)(Pr + kb + 16 + (hi << 3));
    b.q[0] = *(const uint4*)(Hr + (hi << 3));
    b.q[1] = *(const uint4*)(Hr + 16 + (hi << 3));
    acc = __builtin_amdgcn_wmma_f32_16x16x32_bf16(false, a.v, false, b.v, (short)0, acc,
                                                  false, false);
  }

  #pragma unroll
  for (int r = 0; r < 8; ++r) {
    const int m = (hi << 3) + r;            // D layout: VGPR r -> M = r + 8*hi, N = lane&15
    float v = acc[r] * rs[m];
    v = v > 0.f ? v : (__expf(v) - 1.0f);   // ELU (alpha = 1)
    out[((size_t)(bb * NN + i0 + m)) * FOUT + nb + lo] = v;
  }
}

extern "C" void kernel_launch(void* const* d_in, const int* in_sizes, int n_in,
                              void* d_out, int out_size, void* d_ws, size_t ws_size,
                              hipStream_t stream) {
  const float* inp  = (const float*)d_in[0];   // (8,2048,256) f32
  const int*   adj  = (const int*)  d_in[1];   // (8,2048,2048) i32
  const float* Wm   = (const float*)d_in[2];   // (128,256) f32
  const float* bias = (const float*)d_in[3];   // (128,) f32
  const float* avec = (const float*)d_in[4];   // (256,1) f32
  float* out = (float*)d_out;

  char* ws = (char*)d_ws;
  unsigned short* hT = (unsigned short*)ws;                       // 4,194,304 B
  float* ps1 = (float*)(ws + 4194304);                            //   524,288 B
  float* ps2 = (float*)(ws + 4194304 + 524288);                   //   524,288 B
  float* s1  = (float*)(ws + 4194304 + 2 * 524288);               //    65,536 B
  float* s2  = (float*)(ws + 4194304 + 2 * 524288 + 65536);       //    65,536 B

  gat_gemm1 <<<1024, 256, 0, stream>>>(inp, Wm, bias, avec, hT, ps1, ps2);
  gat_scores<<<64,   256, 0, stream>>>(ps1, ps2, s1, s2);
  gat_attn  <<<1024, 256, SMEM_SZ, stream>>>(adj, hT, s1, s2, out);
}
